// CrossAttention_59287728554307
// MI455X (gfx1250) — compile-verified, matrix operand order checked by reading, not run on hardware
//
#include <hip/hip_runtime.h>
#include <hip/hip_bf16.h>

// ---------------------------------------------------------------------------
// CDNA5 (gfx1250) cross-attention, bf16 WMMA pipeline with async LDS staging.
//   B=4, T=2048, T2=1024, C=1024, H=16, D=64
// ---------------------------------------------------------------------------

typedef __attribute__((ext_vector_type(16))) __bf16 v16bf;
typedef __attribute__((ext_vector_type(8)))  float  v8f;

#define CB   1024      // C (embed)
#define TT   2048      // T
#define TT2  1024      // T2
#define BB   4
#define HH   16
#define DD   64

__device__ __forceinline__ unsigned short f2bf(float f) {
    unsigned u = __float_as_uint(f);
    unsigned r = u + 0x7FFFu + ((u >> 16) & 1u);   // round-to-nearest-even
    return (unsigned short)(r >> 16);
}

__device__ __forceinline__ v8f wmma_bf16(v16bf a, v16bf b, v8f c) {
    return __builtin_amdgcn_wmma_f32_16x16x32_bf16(
        /*neg_a=*/false, a, /*neg_b=*/false, b,
        /*c_mod=*/(short)0, c, /*reuse_a=*/false, /*reuse_b=*/false);
}

// Async DMA: 16B global -> LDS, tracked by ASYNCcnt (no VGPR data path).
// LDS byte address = low 32 bits of the generic pointer.
__device__ __forceinline__ void async_copy_b128(void* lds, const void* gptr) {
    unsigned           l = (unsigned)(unsigned long long)lds;
    unsigned long long g = (unsigned long long)gptr;
    asm volatile("global_load_async_to_lds_b128 %0, %1, off"
                 :: "v"(l), "v"(g) : "memory");
}

template <int N>
__device__ __forceinline__ void wait_asynccnt() {
    asm volatile("s_wait_asynccnt %0" :: "i"(N) : "memory");
}

// Load a 16x32 bf16 fragment (A-layout; identical layout serves as B-fragment
// when `base` is the N-major (transposed) matrix).  Per ISA 7.12.2:
//   lane = row%16, half = lane>>4 selects K subgroup, element e<8 -> K=half*8+e,
//   e>=8 -> K=16+half*8+(e-8).  Two 16B LDS loads per fragment.
__device__ __forceinline__ v16bf load_frag(const unsigned short* base, int ld,
                                           int rowBase, int kBase, int lane) {
    int l15 = lane & 15, hl = lane >> 4;
    const unsigned short* p = base + (rowBase + l15) * ld + kBase + hl * 8;
    union { v16bf v; uint4 q[2]; } x;
    x.q[0] = *(const uint4*)(p);
    x.q[1] = *(const uint4*)(p + 16);
    return x.v;
}

// ---------------------------------------------------------------------------
// fp32 -> bf16 elementwise convert
// ---------------------------------------------------------------------------
__global__ __launch_bounds__(256) void cvt_kernel(const float* __restrict__ in,
                                                  unsigned short* __restrict__ out,
                                                  int n) {
    int i = blockIdx.x * 256 + threadIdx.x;
    if (i < n) out[i] = f2bf(in[i]);
}

// W[k][n] (fp32, 1024x1024) -> Wt[n][k] (bf16)
__global__ __launch_bounds__(256) void transpose_cvt_kernel(const float* __restrict__ W,
                                                            unsigned short* __restrict__ Wt) {
    int idx = blockIdx.x * 256 + threadIdx.x;     // over 1M elements
    int n = idx >> 10, k = idx & 1023;
    Wt[idx] = f2bf(W[(size_t)k * CB + n]);
}

// ---------------------------------------------------------------------------
// bf16 GEMM:  C[m][n] = A[m][:] dot Wt[n][:] + bias[n],  N = K = 1024
// Block = 128x128 tile, 8 waves, each wave 32x64 (2x4 WMMA accumulators).
// Double-buffered LDS, tiles filled by global_load_async_to_lds_b128.
// Steady-state loop is branch-free (last iteration peeled).
// mode 0: bf16 row-major out     mode 1: bf16 scatter into Vt[b][h][d][s]
// mode 2: fp32 row-major out (final projection)
// ---------------------------------------------------------------------------
__global__ __launch_bounds__(256)
void gemm_bf16_kernel(const unsigned short* __restrict__ A,
                      const unsigned short* __restrict__ Bt,
                      const float* __restrict__ bias,
                      void* __restrict__ Cout, int mode) {
    constexpr int N = CB, K = CB, KT = K / 32;
    const int n0 = blockIdx.x * 128, m0 = blockIdx.y * 128;
    const int tid = threadIdx.x, lane = tid & 31, wid = tid >> 5;
    const int wm = (wid >> 1) * 32, wn = (wid & 1) * 64;

    __shared__ __align__(16) unsigned short As[2][128 * 32];
    __shared__ __align__(16) unsigned short Bs[2][128 * 32];

    // 4 async b128 copies per thread per tile (2 for A, 2 for B)
    auto issue_tile = [&](int buf, int k0) {
#pragma unroll
        for (int i = 0; i < 2; ++i) {
            int c = tid + i * 256;            // 512 chunks of 8 bf16
            int row = c >> 2, kc = c & 3;
            async_copy_b128(&As[buf][row * 32 + kc * 8],
                            &A[(size_t)(m0 + row) * K + k0 + kc * 8]);
            async_copy_b128(&Bs[buf][row * 32 + kc * 8],
                            &Bt[(size_t)(n0 + row) * K + k0 + kc * 8]);
        }
    };

    v8f acc[2][4] = {};

    auto compute = [&](int buf) {
        v16bf a0 = load_frag(As[buf], 32, wm,      0, lane);
        v16bf a1 = load_frag(As[buf], 32, wm + 16, 0, lane);
#pragma unroll
        for (int nt = 0; nt < 4; ++nt) {
            v16bf bb = load_frag(Bs[buf], 32, wn + nt * 16, 0, lane);
            acc[0][nt] = wmma_bf16(a0, bb, acc[0][nt]);
            acc[1][nt] = wmma_bf16(a1, bb, acc[1][nt]);
        }
    };

    issue_tile(0, 0);
    for (int it = 0; it < KT - 1; ++it) {          // branch-free steady state
        const int buf = it & 1;
        issue_tile(buf ^ 1, (it + 1) * 32);
        wait_asynccnt<4>();
        __syncthreads();                            // all waves' DMA landed
        compute(buf);
        __syncthreads();                            // buffer reusable next-next iter
    }
    wait_asynccnt<0>();                             // peeled last tile
    __syncthreads();
    compute((KT - 1) & 1);

    const int l15 = lane & 15, hl = lane >> 4;
#pragma unroll
    for (int mt = 0; mt < 2; ++mt) {
#pragma unroll
        for (int nt = 0; nt < 4; ++nt) {
            int n = n0 + wn + nt * 16 + l15;
            float bias_n = bias[n];
#pragma unroll
            for (int e = 0; e < 8; ++e) {
                int m = m0 + wm + mt * 16 + e + 8 * hl;
                float val = acc[mt][nt][e] + bias_n;
                if (mode == 0) {
                    ((unsigned short*)Cout)[(size_t)m * N + n] = f2bf(val);
                } else if (mode == 2) {
                    ((float*)Cout)[(size_t)m * N + n] = val;
                } else {   // mode 1: V transposed: Vt[b][h][d][s]
                    int b = m >> 10, s = m & 1023;       // rows are [B*T2]
                    int h = n >> 6,  d = n & 63;
                    ((unsigned short*)Cout)[(((size_t)(b * HH + h)) * DD + d) * TT2 + s] =
                        f2bf(val);
                }
            }
        }
    }
}

// ---------------------------------------------------------------------------
// Flash attention: grid (T/128, B*H).  Block = 256 thr = 8 waves; each wave
// owns a 16-row Q tile, streams K/V in 64-wide double-buffered chunks filled
// by async DMA.  Online softmax, fp32 accumulators.  Last chunk peeled so the
// steady-state loop is branch-free.
// ---------------------------------------------------------------------------
__global__ __launch_bounds__(256)
void flash_attn_kernel(const unsigned short* __restrict__ Qb,
                       const unsigned short* __restrict__ Kb,
                       const unsigned short* __restrict__ Vt,
                       unsigned short* __restrict__ Yb) {
    const int tTile = blockIdx.x;          // 0..15
    const int bh    = blockIdx.y;          // 0..63
    const int b = bh >> 4, h = bh & 15;
    const int tid = threadIdx.x, lane = tid & 31, wid = tid >> 5;
    const int l15 = lane & 15, hl = lane >> 4;
    const int t0 = tTile * 128;

    __shared__ __align__(16) unsigned short Qs[128 * DD];      // 16 KB
    __shared__ __align__(16) unsigned short Ks[2][64 * DD];    // 16 KB
    __shared__ __align__(16) unsigned short Vs[2][DD * 64];    // 16 KB  [d][s]
    __shared__ __align__(16) unsigned short Ps[8][16 * 64];    // 16 KB  per-wave P

    const unsigned short* Qg = Qb + (size_t)b * TT  * CB + h * DD;
    const unsigned short* Kg = Kb + (size_t)b * TT2 * CB + h * DD;
    const unsigned short* Vg = Vt + (size_t)bh * DD * TT2;     // [d][s]

    // Stage Q tile (128 x 64): 4 async copies per thread
#pragma unroll
    for (int i = 0; i < 4; ++i) {
        int c = tid + i * 256;
        int row = c >> 3, kc = c & 7;
        async_copy_b128(&Qs[row * DD + kc * 8],
                        &Qg[(size_t)(t0 + row) * CB + kc * 8]);
    }

    // 4 async copies per thread per s-chunk (2 for K, 2 for V)
    auto issue_kv = [&](int buf, int s0) {
#pragma unroll
        for (int i = 0; i < 2; ++i) {
            int c = tid + i * 256;
            int row = c >> 3, kc = c & 7;
            async_copy_b128(&Ks[buf][row * DD + kc * 8],
                            &Kg[(size_t)(s0 + row) * CB + kc * 8]);
            async_copy_b128(&Vs[buf][row * 64 + kc * 8],
                            &Vg[(size_t)row * TT2 + s0 + kc * 8]);
        }
    };

    v8f   o[4] = {};
    float rm[8], rl[8];
#pragma unroll
    for (int r = 0; r < 8; ++r) { rm[r] = -1e30f; rl[r] = 0.0f; }

    // Per-chunk body: QK^T -> online softmax -> P@V, all on buffer `buf`.
    auto process = [&](int buf) {
        // ---- S = (Q Kt) * 1/sqrt(D):  4 s-tiles x 2 k-steps over D=64 ----
        v8f sacc[4] = {};
        v16bf aq0 = load_frag(Qs, DD, wid * 16, 0,  lane);
        v16bf aq1 = load_frag(Qs, DD, wid * 16, 32, lane);
#pragma unroll
        for (int st = 0; st < 4; ++st) {
            v16bf bk0 = load_frag(Ks[buf], DD, st * 16, 0,  lane);
            v16bf bk1 = load_frag(Ks[buf], DD, st * 16, 32, lane);
            sacc[st] = wmma_bf16(aq0, bk0, sacc[st]);
            sacc[st] = wmma_bf16(aq1, bk1, sacc[st]);
        }
#pragma unroll
        for (int st = 0; st < 4; ++st) sacc[st] = sacc[st] * 0.125f;  // 1/sqrt(64)

        // ---- row stats (rows of one half live in lanes 0..15 / 16..31) ----
        float tmax[8];
#pragma unroll
        for (int r = 0; r < 8; ++r) {
            float mx = sacc[0][r];
#pragma unroll
            for (int st = 1; st < 4; ++st) mx = fmaxf(mx, sacc[st][r]);
#pragma unroll
            for (int off = 1; off < 16; off <<= 1)
                mx = fmaxf(mx, __shfl_xor(mx, off, 32));
            tmax[r] = mx;
        }

        float fac[8], rsum[8];
#pragma unroll
        for (int r = 0; r < 8; ++r) {
            float nm = fmaxf(rm[r], tmax[r]);
            fac[r] = __expf(rm[r] - nm);
            rm[r]  = nm;
            rsum[r] = 0.0f;
        }

        // ---- P = exp(S - m), stage to LDS in A-layout source form ----
#pragma unroll
        for (int st = 0; st < 4; ++st) {
#pragma unroll
            for (int r = 0; r < 8; ++r) {
                float p = __expf(sacc[st][r] - rm[r]);
                rsum[r] += p;
                Ps[wid][(r + 8 * hl) * 64 + st * 16 + l15] = f2bf(p);
            }
        }
#pragma unroll
        for (int r = 0; r < 8; ++r) {
            float s = rsum[r];
#pragma unroll
            for (int off = 1; off < 16; off <<= 1) s += __shfl_xor(s, off, 32);
            rl[r] = rl[r] * fac[r] + s;
        }
#pragma unroll
        for (int dt = 0; dt < 4; ++dt) {
#pragma unroll
            for (int r = 0; r < 8; ++r) o[dt][r] = o[dt][r] * fac[r];
        }

        // ---- O += P @ V :  2 k-steps over s-chunk x 4 d-tiles ----
#pragma unroll
        for (int kk = 0; kk < 64; kk += 32) {
            v16bf ap = load_frag(&Ps[wid][0], 64, 0, kk, lane);
#pragma unroll
            for (int dt = 0; dt < 4; ++dt) {
                v16bf bv_ = load_frag(Vs[buf], 64, dt * 16, kk, lane);
                o[dt] = wmma_bf16(ap, bv_, o[dt]);
            }
        }
    };

    issue_kv(0, 0);
    constexpr int ST = TT2 / 64;           // 16 s-chunks
    for (int it = 0; it < ST - 1; ++it) {  // branch-free steady state
        const int buf = it & 1;
        issue_kv(buf ^ 1, (it + 1) * 64);
        wait_asynccnt<4>();
        __syncthreads();                   // Q + current K/V landed (in-order cnt)
        process(buf);
        __syncthreads();                   // buffer reusable next-next iter
    }
    wait_asynccnt<0>();                    // peeled last chunk
    __syncthreads();
    process((ST - 1) & 1);

    // ---- epilogue: y = O / l ----
    unsigned short* Yg = Yb + (size_t)b * TT * CB + h * DD;
#pragma unroll
    for (int dt = 0; dt < 4; ++dt) {
#pragma unroll
        for (int e = 0; e < 8; ++e) {
            int trow = t0 + wid * 16 + e + 8 * hl;
            int d    = dt * 16 + l15;
            Yg[(size_t)trow * CB + d] = f2bf(o[dt][e] / rl[e]);
        }
    }
}

// ---------------------------------------------------------------------------
// Host launch
// ---------------------------------------------------------------------------
extern "C" void kernel_launch(void* const* d_in, const int* in_sizes, int n_in,
                              void* d_out, int out_size, void* d_ws, size_t ws_size,
                              hipStream_t stream) {
    (void)in_sizes; (void)n_in; (void)out_size; (void)ws_size;
    const float* x   = (const float*)d_in[0];
    const float* enc = (const float*)d_in[1];
    const float* Wq  = (const float*)d_in[2];
    const float* bq  = (const float*)d_in[3];
    const float* Wk  = (const float*)d_in[4];
    const float* bk  = (const float*)d_in[5];
    const float* Wv  = (const float*)d_in[6];
    const float* bv  = (const float*)d_in[7];
    const float* Wo  = (const float*)d_in[8];
    const float* bo  = (const float*)d_in[9];
    float* out = (float*)d_out;

    char* ws = (char*)d_ws;
    size_t off = 0;
    unsigned short* Xb  = (unsigned short*)(ws + off); off += (size_t)BB*TT *CB*2;   // 16 MB
    unsigned short* Eb  = (unsigned short*)(ws + off); off += (size_t)BB*TT2*CB*2;   //  8 MB
    unsigned short* Wqt = (unsigned short*)(ws + off); off += (size_t)CB*CB*2;
    unsigned short* Wkt = (unsigned short*)(ws + off); off += (size_t)CB*CB*2;
    unsigned short* Wvt = (unsigned short*)(ws + off); off += (size_t)CB*CB*2;
    unsigned short* Wot = (unsigned short*)(ws + off); off += (size_t)CB*CB*2;
    unsigned short* Qb  = (unsigned short*)(ws + off); off += (size_t)BB*TT *CB*2;
    unsigned short* Kb  = (unsigned short*)(ws + off); off += (size_t)BB*TT2*CB*2;
    unsigned short* Vtb = (unsigned short*)(ws + off); off += (size_t)BB*TT2*CB*2;
    unsigned short* Yb  = (unsigned short*)(ws + off); off += (size_t)BB*TT *CB*2;

    const int nX = BB * TT * CB;    // 8388608
    const int nE = BB * TT2 * CB;   // 4194304
    cvt_kernel<<<(nX + 255) / 256, 256, 0, stream>>>(x,   Xb, nX);
    cvt_kernel<<<(nE + 255) / 256, 256, 0, stream>>>(enc, Eb, nE);
    transpose_cvt_kernel<<<(CB * CB) / 256, 256, 0, stream>>>(Wq, Wqt);
    transpose_cvt_kernel<<<(CB * CB) / 256, 256, 0, stream>>>(Wk, Wkt);
    transpose_cvt_kernel<<<(CB * CB) / 256, 256, 0, stream>>>(Wv, Wvt);
    transpose_cvt_kernel<<<(CB * CB) / 256, 256, 0, stream>>>(Wo, Wot);

    // Projections
    gemm_bf16_kernel<<<dim3(CB / 128, (BB * TT)  / 128), 256, 0, stream>>>(Xb, Wqt, bq, Qb, 0);
    gemm_bf16_kernel<<<dim3(CB / 128, (BB * TT2) / 128), 256, 0, stream>>>(Eb, Wkt, bk, Kb, 0);
    gemm_bf16_kernel<<<dim3(CB / 128, (BB * TT2) / 128), 256, 0, stream>>>(Eb, Wvt, bv, Vtb, 1);

    // Attention
    flash_attn_kernel<<<dim3(TT / 128, BB * HH), 256, 0, stream>>>(Qb, Kb, Vtb, Yb);

    // Output projection (fp32 out + bias)
    gemm_bf16_kernel<<<dim3(CB / 128, (BB * TT) / 128), 256, 0, stream>>>(Yb, Wot, bo, out, 2);
}
